// MultiHeadAttentionLayer_85194971283944
// MI455X (gfx1250) — matrix operand order repaired and checked by reading, not compiled, not run on hardware
//
#include <hip/hip_runtime.h>

// ---------- CDNA5 WMMA types ----------
typedef __attribute__((ext_vector_type(16))) __bf16        v16bf;
typedef __attribute__((ext_vector_type(8)))  float         v8f;
typedef __attribute__((ext_vector_type(8)))  unsigned int  v8u;

#define HIDDEN 1024
#define SEQ    2048
#define NBATCH 2
#define HEADS  16
#define HDIM   64
#define ROWS   (NBATCH * SEQ) // 4096

// ---------- helpers ----------
__device__ __forceinline__ unsigned short f2bf(float f) {
  unsigned int u = __float_as_uint(f);
  u += 0x7fffu + ((u >> 16) & 1u);           // round-to-nearest-even
  return (unsigned short)(u >> 16);
}
__device__ __forceinline__ unsigned int pack2bf(float lo, float hi) {
  return (unsigned int)f2bf(lo) | ((unsigned int)f2bf(hi) << 16);
}
__device__ __forceinline__ v16bf make16(uint4 a, uint4 b) {
  v8u t;
  t[0]=a.x; t[1]=a.y; t[2]=a.z; t[3]=a.w;
  t[4]=b.x; t[5]=b.y; t[6]=b.z; t[7]=b.w;
  return __builtin_bit_cast(v16bf, t);
}
__device__ __forceinline__ v8f wmma_bf16(v16bf a, v16bf b, v8f c) {
  return __builtin_amdgcn_wmma_f32_16x16x32_bf16(false, a, false, b,
                                                 (short)0, c, false, false);
}

// ---------- 1) LayerNorm (fp32 -> bf16) ----------
__global__ void __launch_bounds__(256)
ln_bf16_kernel(const float* __restrict__ x, const float* __restrict__ g,
               const float* __restrict__ bb, unsigned short* __restrict__ xn) {
  const int row = blockIdx.x;
  const int tid = threadIdx.x;                 // 256 threads, 4 elems each
  const float4 v = ((const float4*)(x + (long)row * HIDDEN))[tid];
  float s  = v.x + v.y + v.z + v.w;
  float s2 = v.x*v.x + v.y*v.y + v.z*v.z + v.w*v.w;
#pragma unroll
  for (int off = 16; off; off >>= 1) {
    s  += __shfl_xor(s,  off, 32);
    s2 += __shfl_xor(s2, off, 32);
  }
  __shared__ float ps[8], ps2[8];
  const int w = tid >> 5;
  if ((tid & 31) == 0) { ps[w] = s; ps2[w] = s2; }
  __syncthreads();
  s = 0.f; s2 = 0.f;
#pragma unroll
  for (int i = 0; i < 8; ++i) { s += ps[i]; s2 += ps2[i]; }
  const float mu   = s * (1.0f / HIDDEN);
  const float var  = s2 * (1.0f / HIDDEN) - mu * mu;
  const float rstd = rsqrtf(var + 1e-5f);
  const float4 gv = ((const float4*)g)[tid];
  const float4 bv = ((const float4*)bb)[tid];
  const float y0 = (v.x - mu) * rstd * gv.x + bv.x;
  const float y1 = (v.y - mu) * rstd * gv.y + bv.y;
  const float y2 = (v.z - mu) * rstd * gv.z + bv.z;
  const float y3 = (v.w - mu) * rstd * gv.w + bv.w;
  uint2 o; o.x = pack2bf(y0, y1); o.y = pack2bf(y2, y3);
  ((uint2*)(xn + (long)row * HIDDEN))[tid] = o;
}

// ---------- 2) weight transpose + cast: wt[n][k] = bf16(w[k][n]) ----------
__global__ void __launch_bounds__(256)
wtranspose_kernel(const float* __restrict__ w, unsigned short* __restrict__ wt) {
  __shared__ float tile[32][33];
  const int x0 = blockIdx.x * 32;   // n
  const int y0 = blockIdx.y * 32;   // k
#pragma unroll
  for (int i = threadIdx.y; i < 32; i += 8)
    tile[i][threadIdx.x] = w[(long)(y0 + i) * HIDDEN + x0 + threadIdx.x];
  __syncthreads();
#pragma unroll
  for (int i = threadIdx.y; i < 32; i += 8)
    wt[(long)(x0 + i) * HIDDEN + y0 + threadIdx.x] = f2bf(tile[threadIdx.x][i]);
}

// ---------- 3) WMMA GEMM: D[m][n] = A[m][:]·Wt[n][:] + bias[n] ----------
// A: bf16 [M][K] row-major.  Wt: bf16 [N][K] (pre-transposed weight).
// mode 0: out bf16 at [(b*16+h)][s][dd]     (Q/K layout, s-major)
// mode 1: out bf16 at [(b*16+h)][dd][s]     (V transposed)
// mode 2: out fp32 = val + resid            (final projection + residual)
__global__ void __launch_bounds__(256)
gemm_wmma_kernel(const unsigned short* __restrict__ A,
                 const unsigned short* __restrict__ Wt,
                 const float* __restrict__ bias,
                 void* __restrict__ outp,
                 const float* __restrict__ resid,
                 int M, int N, int K, int mode) {
  const int lane = threadIdx.x & 31;
  const int wave = blockIdx.x * (blockDim.x >> 5) + (threadIdx.x >> 5);
  const int tilesN = N >> 6;
  const int tm = wave / tilesN, tn = wave % tilesN;
  if (tm * 64 >= M) return;
  const int m0 = tm * 64, n0 = tn * 64;
  const int half = lane >> 4, l16 = lane & 15;

  v8f c[4][4];
#pragma unroll
  for (int i = 0; i < 4; ++i)
#pragma unroll
    for (int j = 0; j < 4; ++j)
      c[i][j] = (v8f){0.f,0.f,0.f,0.f,0.f,0.f,0.f,0.f};

  for (int kk = 0; kk < K; kk += 32) {
    v16bf a[4];
#pragma unroll
    for (int ti = 0; ti < 4; ++ti) {  // A: lane l16 = row; VGPR0-3 K=half*8..+7, VGPR4-7 K=16+half*8..+7
      const unsigned short* p = A + (long)(m0 + ti * 16 + l16) * K + kk + half * 8;
      a[ti] = make16(*(const uint4*)p, *(const uint4*)(p + 16));
    }
    if (kk + 64 < K) {  // gfx1250 global_prefetch hint
      __builtin_prefetch(A  + (long)(m0 + l16) * K + kk + 64, 0, 1);
      __builtin_prefetch(Wt + (long)(n0 + l16) * K + kk + 64, 0, 1);
    }
#pragma unroll
    for (int tj = 0; tj < 4; ++tj) {  // B: lane l16 = col; VGPR0-3 K=half*8..+7, VGPR4-7 K=16+half*8..+7
      const unsigned short* p = Wt + (long)(n0 + tj * 16 + l16) * K + kk + half * 8;
      v16bf bm = make16(*(const uint4*)p, *(const uint4*)(p + 16));
#pragma unroll
      for (int ti = 0; ti < 4; ++ti)
        c[ti][tj] = wmma_bf16(a[ti], bm, c[ti][tj]);
    }
  }

  // epilogue: C tile -> lane = col (l16), rows r + half*8
#pragma unroll
  for (int tj = 0; tj < 4; ++tj) {
    const int n_g = n0 + tj * 16 + l16;
    const float bv = bias[n_g];
#pragma unroll
    for (int ti = 0; ti < 4; ++ti) {
#pragma unroll
      for (int r = 0; r < 8; ++r) {
        const int m_g = m0 + ti * 16 + r + half * 8;
        const float val = c[ti][tj][r] + bv;
        if (mode == 2) {
          ((float*)outp)[(long)m_g * N + n_g] = val + resid[(long)m_g * N + n_g];
        } else {
          const int b = m_g >> 11, s = m_g & (SEQ - 1);
          const int h = n_g >> 6,  dd = n_g & (HDIM - 1);
          long off;
          if (mode == 0) off = (long)(b * HEADS + h) * SEQ * HDIM + (long)s * HDIM + dd;
          else           off = (long)(b * HEADS + h) * SEQ * HDIM + (long)dd * SEQ + s;
          ((unsigned short*)outp)[off] = f2bf(val);
        }
      }
    }
  }
}

// ---------- 4) Flash attention, per-wave 16-query tile, all-register ----------
// q,k: bf16 [BH][S][d].  vt: bf16 [BH][d][S].  attn out: bf16 [B][S][H*d].
// Computes S^T = K·Q^T so softmax stats are per-lane (lane = query column),
// then O^T = V^T·P^T.  One shfl_xor(16) merges lane halves.
__global__ void __launch_bounds__(256)
attn_wmma_kernel(const unsigned short* __restrict__ q,
                 const unsigned short* __restrict__ k,
                 const unsigned short* __restrict__ vt,
                 unsigned short* __restrict__ attn) {
  const int lane = threadIdx.x & 31;
  const int wave = blockIdx.x * 8 + (threadIdx.x >> 5);   // 4096 waves total
  const int bh = wave >> 7;                               // 32 (b,h) planes
  const int q0 = (wave & 127) << 4;                       // 16-query tile
  const int half = lane >> 4, l16 = lane & 15;
  const unsigned short* qb = q  + (long)bh * SEQ * HDIM;
  const unsigned short* kb = k  + (long)bh * SEQ * HDIM;
  const unsigned short* vb = vt + (long)bh * HDIM * SEQ;

  // Q^T as two B-matrices (lane = query col, K-dim = d contiguous)
  v16bf Bq0, Bq1;
  {
    const unsigned short* p = qb + (long)(q0 + l16) * HDIM + half * 8;
    Bq0 = make16(*(const uint4*)p,        *(const uint4*)(p + 16));  // d 0..31
    Bq1 = make16(*(const uint4*)(p + 32), *(const uint4*)(p + 48));  // d 32..63
  }

  float m_i = -3e38f, l_i = 0.f;
  v8f o[4];
#pragma unroll
  for (int t = 0; t < 4; ++t) o[t] = (v8f){0.f,0.f,0.f,0.f,0.f,0.f,0.f,0.f};

  for (int k0 = 0; k0 < SEQ; k0 += 32) {
    // scores^T: two 16(key)x16(query) tiles
    v8f sc[2];
#pragma unroll
    for (int h2 = 0; h2 < 2; ++h2) {
      const unsigned short* p = kb + (long)(k0 + h2 * 16 + l16) * HDIM + half * 8;
      v16bf A0 = make16(*(const uint4*)p,        *(const uint4*)(p + 16)); // d 0..31
      v16bf A1 = make16(*(const uint4*)(p + 32), *(const uint4*)(p + 48)); // d 32..63
      v8f z = (v8f){0.f,0.f,0.f,0.f,0.f,0.f,0.f,0.f};
      z = wmma_bf16(A0, Bq0, z);
      sc[h2] = wmma_bf16(A1, Bq1, z);
    }
    // online softmax, per-lane (lane owns query l16; partner lane^16 has other keys)
    float ss[16];
#pragma unroll
    for (int h2 = 0; h2 < 2; ++h2)
#pragma unroll
      for (int r = 0; r < 8; ++r) ss[h2 * 8 + r] = sc[h2][r] * 0.125f; // 1/sqrt(64)
    float mloc = ss[0];
#pragma unroll
    for (int i = 1; i < 16; ++i) mloc = fmaxf(mloc, ss[i]);
    mloc = fmaxf(mloc, __shfl_xor(mloc, 16, 32));
    const float m_new = fmaxf(m_i, mloc);
    const float alpha = __expf(m_i - m_new);
    float pv[16], sum = 0.f;
#pragma unroll
    for (int i = 0; i < 16; ++i) { pv[i] = __expf(ss[i] - m_new); sum += pv[i]; }
    sum += __shfl_xor(sum, 16, 32);
    l_i = l_i * alpha + sum;
    m_i = m_new;
#pragma unroll
    for (int t = 0; t < 4; ++t)
#pragma unroll
      for (int r = 0; r < 8; ++r) o[t][r] *= alpha;

    // rebuild P^T (32 keys x 16 queries) into B-matrix layout: one half-swap shuffle
    float rec[8];
#pragma unroll
    for (int r = 0; r < 8; ++r) {
      const float sel = half ? pv[r] : pv[8 + r];
      rec[r] = __shfl_xor(sel, 16, 32);
    }
    float lo8[8], hi8[8];
#pragma unroll
    for (int r = 0; r < 8; ++r) {
      lo8[r] = half ? rec[r]     : pv[r];      // keys base+0..7 of this lane-half
      hi8[r] = half ? pv[8 + r]  : rec[r];     // keys base+8..15
    }
    v8u pb;
#pragma unroll
    for (int j = 0; j < 4; ++j) pb[j]     = pack2bf(lo8[2*j], lo8[2*j+1]);
#pragma unroll
    for (int j = 0; j < 4; ++j) pb[4 + j] = pack2bf(hi8[2*j], hi8[2*j+1]);
    const v16bf Bp = __builtin_bit_cast(v16bf, pb);

    // O^T += V^T_tile(16d x 32k) * P^T(32k x 16q)
#pragma unroll
    for (int t = 0; t < 4; ++t) {
      const unsigned short* p = vb + (long)(t * 16 + l16) * SEQ + k0 + half * 16;
      v16bf Av = make16(*(const uint4*)p, *(const uint4*)(p + 8));
      o[t] = wmma_bf16(Av, Bp, o[t]);
    }
  }

  // normalize and store: lane owns query s = q0+l16; rows are d indices
  const float inv = 1.0f / l_i;
  const int b = bh >> 4, h = bh & (HEADS - 1);
  const int s = q0 + l16;
  unsigned short* op = attn + ((long)b * SEQ + s) * HIDDEN + h * HDIM;
#pragma unroll
  for (int t = 0; t < 4; ++t)
#pragma unroll
    for (int r = 0; r < 8; ++r)
      op[t * 16 + r + half * 8] = f2bf(o[t][r] * inv);
}

// ---------- host launcher ----------
extern "C" void kernel_launch(void* const* d_in, const int* in_sizes, int n_in,
                              void* d_out, int out_size, void* d_ws, size_t ws_size,
                              hipStream_t stream) {
  (void)in_sizes; (void)n_in; (void)out_size; (void)ws_size;
  const float* x    = (const float*)d_in[0];
  const float* wq   = (const float*)d_in[1];
  const float* bq   = (const float*)d_in[2];
  const float* wk   = (const float*)d_in[3];
  const float* bk   = (const float*)d_in[4];
  const float* wv   = (const float*)d_in[5];
  const float* bv   = (const float*)d_in[6];
  const float* wo   = (const float*)d_in[7];
  const float* bo   = (const float*)d_in[8];
  const float* ln_g = (const float*)d_in[9];
  const float* ln_b = (const float*)d_in[10];

  // workspace partition (40 MB total)
  char* ws = (char*)d_ws;
  const size_t MB = 1u << 20;
  unsigned short* xn  = (unsigned short*)(ws + 0);        // 8 MB, reused as attn-out
  unsigned short* wqt = (unsigned short*)(ws + 8  * MB);  // 2 MB each
  unsigned short* wkt = (unsigned short*)(ws + 10 * MB);
  unsigned short* wvt = (unsigned short*)(ws + 12 * MB);
  unsigned short* wot = (unsigned short*)(ws + 14 * MB);
  unsigned short* qb  = (unsigned short*)(ws + 16 * MB);  // 8 MB  [BH][S][d]
  unsigned short* kb  = (unsigned short*)(ws + 24 * MB);  // 8 MB  [BH][S][d]
  unsigned short* vtb = (unsigned short*)(ws + 32 * MB);  // 8 MB  [BH][d][S]
  unsigned short* attn = xn;                              // alias (xn dead after QKV)

  // 1) layernorm -> bf16
  ln_bf16_kernel<<<ROWS, 256, 0, stream>>>(x, ln_g, ln_b, xn);

  // 2) transpose+cast the four weights
  dim3 tb(32, 8), tg(HIDDEN / 32, HIDDEN / 32);
  wtranspose_kernel<<<tg, tb, 0, stream>>>(wq, wqt);
  wtranspose_kernel<<<tg, tb, 0, stream>>>(wk, wkt);
  wtranspose_kernel<<<tg, tb, 0, stream>>>(wv, wvt);
  wtranspose_kernel<<<tg, tb, 0, stream>>>(wo, wot);

  // 3) QKV projections (WMMA): 64x64 tiles -> 1024 waves -> 128 blocks
  const int gemm_blocks = (ROWS / 64) * (HIDDEN / 64) / 8;
  gemm_wmma_kernel<<<gemm_blocks, 256, 0, stream>>>(xn, wqt, bq, (void*)qb,  nullptr, ROWS, HIDDEN, HIDDEN, 0);
  gemm_wmma_kernel<<<gemm_blocks, 256, 0, stream>>>(xn, wkt, bk, (void*)kb,  nullptr, ROWS, HIDDEN, HIDDEN, 0);
  gemm_wmma_kernel<<<gemm_blocks, 256, 0, stream>>>(xn, wvt, bv, (void*)vtb, nullptr, ROWS, HIDDEN, HIDDEN, 1);

  // 4) attention: 32 planes * 128 q-tiles = 4096 waves -> 512 blocks
  attn_wmma_kernel<<<512, 256, 0, stream>>>(qb, kb, vtb, attn);

  // 5) output projection + bias + residual -> fp32 d_out
  gemm_wmma_kernel<<<gemm_blocks, 256, 0, stream>>>(attn, wot, bo, d_out, x, ROWS, HIDDEN, HIDDEN, 2);
}
